// GAE_11785390260515
// MI455X (gfx1250) — matrix operand order repaired and checked by reading, not compiled, not run on hardware
//
#include <hip/hip_runtime.h>
#include <hip/hip_bf16.h>
#include <math.h>

typedef __attribute__((ext_vector_type(16))) __bf16 v16bf;
typedef __attribute__((ext_vector_type(8)))  float  v8f;

#define NC   5
#define DIM  64
#define H0T  64
#define H1T  32
#define BU   2048
#define BV   2048

__device__ __forceinline__ unsigned short f2bf(float f) {
  unsigned int x = __float_as_uint(f);
  x += 0x7FFFu + ((x >> 16) & 1u);   // round-to-nearest-even
  return (unsigned short)(x >> 16);
}

// ---------------------------------------------------------------- K0: init
__global__ void init_kernel(float* accums) {
  accums[0] = 0.f; accums[1] = 0.f; accums[2] = 0.f;
}

// ------------------------------------------------- K1: gather ratings -> code
__global__ __launch_bounds__(256) void build_code_kernel(
    const int* __restrict__ u, const int* __restrict__ v,
    const float* __restrict__ ratings,
    unsigned char* __restrict__ code, unsigned char* __restrict__ codeT) {
  int j = blockIdx.x * 256 + threadIdx.x;
  int i = blockIdx.y;
  int uu = u[i], vv = v[j];
  const float* base = ratings + (size_t)uu * 4096 + vv;
  unsigned char cb = 7;
#pragma unroll
  for (int c = 0; c < NC; ++c)
    if (base[(size_t)c * 16777216] > 0.5f) cb = (unsigned char)c;
  code[(size_t)i * BV + j] = cb;
  codeT[(size_t)j * BU + i] = cb;
}

// ---------------------------------------------- K1b: degree counts / scales
__global__ void counts_kernel(const unsigned char* __restrict__ code,
                              const unsigned char* __restrict__ codeT,
                              float* __restrict__ scu, float* __restrict__ scv,
                              float* __restrict__ accums) {
  int t = blockIdx.x * 256 + threadIdx.x;  // 0..4095
  bool isU = t < BU;                       // c[0:BU]=du (column sums), c[BU:]=di (row sums)
  int r = isU ? t : t - BU;
  const unsigned long long* p =
      (const unsigned long long*)((isU ? codeT : code) + (size_t)r * 2048);
  int cnt = 0;
  for (int k = 0; k < 256; ++k) {
    unsigned long long w = p[k];
#pragma unroll
    for (int b = 0; b < 8; ++b) cnt += (int)(((w >> (8 * b)) & 0xFFull) < 5ull);
  }
  float inv = cnt > 0 ? 1.0f / (float)cnt : 0.0f;
  if (isU) { scu[r] = inv; atomicAdd(&accums[0], (float)cnt); }
  else     { scv[r] = inv; }
}

// -------------------------------- K2: support = x @ gc_w  (stacked bf16 B)
__global__ __launch_bounds__(256) void support_kernel(
    const int* __restrict__ u, const int* __restrict__ v,
    const float* __restrict__ u_emb, const float* __restrict__ v_emb,
    const float* __restrict__ gc_w,
    unsigned short* __restrict__ sus, unsigned short* __restrict__ svs) {
  int idx = blockIdx.x * 256 + threadIdx.x;  // < 5*4096*64
  int h = idx & 63;
  int n = (idx >> 6) & 4095;
  int c = idx >> 18;
  const float* x = (n < BU) ? (u_emb + (size_t)u[n] * DIM)
                            : (v_emb + (size_t)v[n - BU] * DIM);
  const float* w = gc_w + (size_t)c * 4096 + h;
  float s = 0.f;
#pragma unroll 8
  for (int d = 0; d < DIM; ++d) s += x[d] * w[(size_t)d * 64];
  unsigned short bits = f2bf(s);
  if (n < BU) sus[((size_t)c * BU + n) * 64 + h]        = bits;
  else        svs[((size_t)c * BV + (n - BU)) * 64 + h] = bits;
}

// -- K3: per-class partial aggsum[c] = onehot(code, c) @ Bstack[c]
//    grid = (16, 5): blockIdx.y = class slab (K split for occupancy)
__global__ __launch_bounds__(256) void agg_wmma_kernel(
    const unsigned char*  __restrict__ code,    // [2048, 2048] A rows
    const unsigned short* __restrict__ Bstack,  // [5*2048, 64] bf16 bits
    float* __restrict__ outPart) {              // [5, 2048, 64] f32 partials
  __shared__ unsigned short bs[32 * 64];        // B chunk, stored transposed [n][k]
  const int cls  = blockIdx.y;                  // class handled by this block
  const int lane = threadIdx.x & 31;
  const int wave = threadIdx.x >> 5;
  const int lrow = lane & 15;
  const int hi   = lane >> 4;
  const int m0   = (blockIdx.x * 8 + wave) * 16;
  const int t  = threadIdx.x;
  const int br = t >> 3;          // 0..31 (k row in chunk)
  const int bc = (t & 7) << 3;    // 0..56 (n col base)
  const unsigned short* Bcls = Bstack + (size_t)cls * 2048 * 64;

  v8f acc[4] = {};
  for (int kk = 0; kk < 2048; kk += 32) {
    __syncthreads();
    {   // cooperative load of 32x64 B chunk, transposed into LDS
      unsigned short tmp[8];
      *(uint4*)tmp = *(const uint4*)(Bcls + (size_t)(kk + br) * 64 + bc);
#pragma unroll
      for (int x = 0; x < 8; ++x) bs[(bc + x) * 32 + br] = tmp[x];
    }
    __syncthreads();

    // A fragment: expand code bytes -> 0/1 bf16 (layout: k = hi*8+e | 16+hi*8+e)
    const unsigned char* cp = code + (size_t)(m0 + lrow) * 2048 + kk + hi * 8;
    unsigned long long w0 = *(const unsigned long long*)cp;
    unsigned long long w1 = *(const unsigned long long*)(cp + 16);
    __builtin_prefetch(cp + 64, 0, 0);          // code bytes two chunks ahead
    union { v16bf v; unsigned short s[16]; } a;
#pragma unroll
    for (int e = 0; e < 8; ++e) {
      a.s[e]     = ((int)((w0 >> (8 * e)) & 0xFF) == cls) ? 0x3F80u : 0u;
      a.s[e + 8] = ((int)((w1 >> (8 * e)) & 0xFF) == cls) ? 0x3F80u : 0u;
    }

    // Preload all four B fragments into distinct registers so the eight
    // ds_load_b128 issue as one clause, then the four WMMAs go back-to-back.
    union BU_t { v16bf v; unsigned short s[16]; uint4 q[2]; } bfr[4];
#pragma unroll
    for (int ht = 0; ht < 4; ++ht) {
      const unsigned short* lp = &bs[(ht * 16 + lrow) * 32 + hi * 8];
      bfr[ht].q[0] = *(const uint4*)lp;
      bfr[ht].q[1] = *(const uint4*)(lp + 16);
    }
#pragma unroll
    for (int ht = 0; ht < 4; ++ht)
      acc[ht] = __builtin_amdgcn_wmma_f32_16x16x32_bf16(
          false, a.v, false, bfr[ht].v, (short)0, acc[ht], false, false);
  }
  float* op = outPart + (size_t)cls * 2048 * 64;
#pragma unroll
  for (int ht = 0; ht < 4; ++ht)
#pragma unroll
    for (int g = 0; g < 8; ++g)
      op[(size_t)(m0 + hi * 8 + g) * 64 + ht * 16 + lrow] = acc[ht][g];
}

// --------------------- K4: z = relu(scale*sum_c agg + bias); hidden = sigmoid(z@W)
__global__ __launch_bounds__(64) void hidden_kernel(
    const float* __restrict__ aggu, const float* __restrict__ aggv,
    const float* __restrict__ scu, const float* __restrict__ scv,
    const float* __restrict__ gc_b, const float* __restrict__ dense_w,
    const float* __restrict__ dense_b, float* __restrict__ hidden) {
  __shared__ float zs[64];
  const int n = blockIdx.x;    // 0..4095
  const int t = threadIdx.x;   // 0..63
  const float* part;
  float sc;
  if (n < BU) { part = aggu + (size_t)n * 64 + t;        sc = scu[n]; }
  else        { part = aggv + (size_t)(n - BU) * 64 + t; sc = scv[n - BU]; }
  float agg = 0.f;
#pragma unroll
  for (int c = 0; c < NC; ++c) agg += part[(size_t)c * 2048 * 64];
  float bsum = 0.f;
#pragma unroll
  for (int c = 0; c < NC; ++c) bsum += gc_b[c * 64 + t];
  zs[t] = fmaxf(sc * agg + bsum, 0.f);
  __syncthreads();
  if (t < H1T) {
    float s = dense_b[t];
#pragma unroll 8
    for (int d = 0; d < H0T; ++d) s += zs[d] * dense_w[d * H1T + t];
    hidden[(size_t)n * H1T + t] = 1.f / (1.f + expf(-s));
  }
}

// ----------------- K5: G[i, c*32+e] = hu @ dec_w[c]; hv -> bf16
__global__ __launch_bounds__(256) void proj_kernel(
    const float* __restrict__ hidden, const float* __restrict__ dec_w,
    unsigned short* __restrict__ G, unsigned short* __restrict__ hvb) {
  int idx = blockIdx.x * 256 + threadIdx.x;  // 0 .. 393215
  if (idx < BU * 160) {
    int e = idx & 31;
    int ic = idx >> 5;          // i*5 + c
    int c = ic % 5, i = ic / 5;
    const float* hu = hidden + (size_t)i * H1T;
    const float* w  = dec_w + (size_t)c * 1024 + e;
    float s = 0.f;
#pragma unroll
    for (int d = 0; d < H1T; ++d) s += hu[d] * w[d * H1T];
    G[(size_t)i * 160 + c * 32 + e] = f2bf(s);
  } else {
    int r = idx - BU * 160;     // 0..65535
    hvb[r] = f2bf(hidden[(size_t)BU * H1T + r]);
  }
}

// ------ K6: logits (5 WMMA per tile), softmax, m_hat, loss & accuracy
__global__ __launch_bounds__(256) void decode_kernel(
    const unsigned short* __restrict__ G,    // [2048, 160] bf16
    const unsigned short* __restrict__ hvb,  // [2048, 32]  bf16
    const unsigned char*  __restrict__ code, // [2048, 2048]
    float* __restrict__ mhat,                // [2048, 2048]
    float* __restrict__ accums) {            // [0]=nobs [1]=loss [2]=acc
  const int lane = threadIdx.x & 31;
  const int wave = threadIdx.x >> 5;
  const int tile = blockIdx.x * 8 + wave;    // 0..16383
  const int i0 = (tile >> 7) * 16;
  const int j0 = (tile & 127) * 16;
  const int lrow = lane & 15;
  const int hi   = lane >> 4;

  union { v16bf v; unsigned short s[16]; uint4 q[2]; } b;
  {
    const unsigned short* hp = hvb + (size_t)(j0 + lrow) * 32 + hi * 8;
    b.q[0] = *(const uint4*)hp;
    b.q[1] = *(const uint4*)(hp + 16);
  }
  v8f acc[NC] = {};
  const unsigned short* gp = G + (size_t)(i0 + lrow) * 160 + hi * 8;
#pragma unroll
  for (int c = 0; c < NC; ++c) {
    union { v16bf v; unsigned short s[16]; uint4 q[2]; } a;
    a.q[0] = *(const uint4*)(gp + c * 32);
    a.q[1] = *(const uint4*)(gp + c * 32 + 16);
    acc[c] = __builtin_amdgcn_wmma_f32_16x16x32_bf16(
        false, a.v, false, b.v, (short)0, acc[c], false, false);
  }

  float lossSum = 0.f, accSum = 0.f;
#pragma unroll
  for (int g = 0; g < 8; ++g) {
    float lg[NC];
#pragma unroll
    for (int c = 0; c < NC; ++c) lg[c] = acc[c][g];
    float best = lg[0]; int pred = 0;
#pragma unroll
    for (int c = 1; c < NC; ++c)
      if (lg[c] > best) { best = lg[c]; pred = c; }  // first-max (jnp.argmax)
    float es = 0.f, num = 0.f;
#pragma unroll
    for (int c = 0; c < NC; ++c) {
      float ex = expf(lg[c] - best);
      es += ex; num += (float)c * ex;
    }
    const int row = i0 + hi * 8 + g;
    const int col = j0 + lrow;
    mhat[(size_t)row * 2048 + col] = num / es;
    unsigned char cb = code[(size_t)row * 2048 + col];
    if (cb < NC) {
      float lcb = lg[0];
#pragma unroll
      for (int c = 1; c < NC; ++c) lcb = ((int)cb == c) ? lg[c] : lcb;
      lossSum += (logf(es) + best - lcb);          // -logp[true]
      accSum  += (pred == (int)cb) ? 1.f : 0.f;
    }
  }
#pragma unroll
  for (int off = 16; off > 0; off >>= 1) {
    lossSum += __shfl_xor(lossSum, off, 32);
    accSum  += __shfl_xor(accSum,  off, 32);
  }
  if (lane == 0) {
    atomicAdd(&accums[1], lossSum);
    atomicAdd(&accums[2], accSum);
  }
}

// ---------------------------------------------------------------- K7: final
__global__ void final_kernel(const float* __restrict__ accums,
                             float* __restrict__ out_tail) {
  float n = fmaxf(accums[0], 1.0f);
  out_tail[0] = accums[1] / n;
  out_tail[1] = accums[2] / n;
}

extern "C" void kernel_launch(void* const* d_in, const int* in_sizes, int n_in,
                              void* d_out, int out_size, void* d_ws, size_t ws_size,
                              hipStream_t stream) {
  (void)in_sizes; (void)n_in; (void)out_size; (void)ws_size;
  const int*   u       = (const int*)d_in[0];
  const int*   v       = (const int*)d_in[1];
  const float* ratings = (const float*)d_in[2];
  const float* u_emb   = (const float*)d_in[3];
  const float* v_emb   = (const float*)d_in[4];
  const float* gc_w    = (const float*)d_in[5];
  const float* gc_b    = (const float*)d_in[6];
  const float* dense_w = (const float*)d_in[7];
  const float* dense_b = (const float*)d_in[8];
  const float* dec_w   = (const float*)d_in[9];
  float* mhat = (float*)d_out;
  float* tail = mhat + (size_t)2048 * 2048;

  char* wsp = (char*)d_ws;
  size_t o = 0;
  auto take = [&](size_t bytes) {
    char* p = wsp + o; o += (bytes + 255) & ~(size_t)255; return p;
  };
  unsigned char*  code  = (unsigned char*)take((size_t)2048 * 2048);
  unsigned char*  codeT = (unsigned char*)take((size_t)2048 * 2048);
  unsigned short* sus   = (unsigned short*)take((size_t)5 * 2048 * 64 * 2);
  unsigned short* svs   = (unsigned short*)take((size_t)5 * 2048 * 64 * 2);
  float*          aggu  = (float*)take((size_t)5 * 2048 * 64 * 4);  // per-class partials
  float*          aggv  = (float*)take((size_t)5 * 2048 * 64 * 4);
  float*          scu   = (float*)take(2048 * 4);
  float*          scv   = (float*)take(2048 * 4);
  float*          hidden= (float*)take((size_t)4096 * 32 * 4);
  unsigned short* G     = (unsigned short*)take((size_t)2048 * 160 * 2);
  unsigned short* hvb   = (unsigned short*)take((size_t)2048 * 32 * 2);
  float*          accums= (float*)take(256);

  init_kernel<<<1, 1, 0, stream>>>(accums);
  build_code_kernel<<<dim3(8, 2048), 256, 0, stream>>>(u, v, ratings, code, codeT);
  counts_kernel<<<16, 256, 0, stream>>>(code, codeT, scu, scv, accums);
  support_kernel<<<5120, 256, 0, stream>>>(u, v, u_emb, v_emb, gc_w, sus, svs);
  agg_wmma_kernel<<<dim3(16, 5), 256, 0, stream>>>(code,  svs, aggu);  // agg_u partials
  agg_wmma_kernel<<<dim3(16, 5), 256, 0, stream>>>(codeT, sus, aggv);  // agg_v partials
  hidden_kernel<<<4096, 64, 0, stream>>>(aggu, aggv, scu, scv, gc_b,
                                         dense_w, dense_b, hidden);
  proj_kernel<<<1536, 256, 0, stream>>>(hidden, dec_w, G, hvb);
  decode_kernel<<<2048, 256, 0, stream>>>(G, hvb, code, mhat, accums);
  final_kernel<<<1, 1, 0, stream>>>(accums, tail);
}
